// Neighborhood2DAttention_30477087932850
// MI455X (gfx1250) — matrix operand order repaired and checked by reading, not compiled, not run on hardware
//
#include <hip/hip_runtime.h>

// ---------------------------------------------------------------------------
// Types for CDNA5 WMMA / TDM (wave32)
// ---------------------------------------------------------------------------
typedef __attribute__((ext_vector_type(16))) __bf16 v16bf;
typedef __attribute__((ext_vector_type(8)))  __bf16 v8bf;
typedef __attribute__((ext_vector_type(8)))  float  v8f;
typedef __attribute__((ext_vector_type(4)))  unsigned int u32x4;
typedef __attribute__((ext_vector_type(8)))  int    i32x8;
typedef __attribute__((ext_vector_type(4)))  int    i32x4;

__device__ __forceinline__ int imin(int a, int b) { return a < b ? a : b; }
__device__ __forceinline__ int imax(int a, int b) { return a > b ? a : b; }

__device__ __forceinline__ v8f wmma_bf16(v16bf a, v16bf b, v8f c) {
  // (neg_a, A, neg_b, B, c_mod, C, reuse_a, reuse_b)
  return __builtin_amdgcn_wmma_f32_16x16x32_bf16(false, a, false, b,
                                                 (short)0, c, false, false);
}

// Problem constants
#define QLEN 4096
#define HID  1024
#define NH   16
#define DH   64
#define XD   64
#define YD   64

// ---------------------------------------------------------------------------
// TDM: async-DMA a 2D bf16 tile (tile_dim0=32 elems contiguous, tile_dim1=64
// rows, row stride = K elems) from global into LDS. D# packing per
// CDNA5 ISA. Tracked by TENSORcnt.
// This toolchain exposes the 6-arg builtin:
//   (u32x4 g0, i32x8 g1, i32x4 g2, i32x4 g3, i32x8, i32 cpol)
// ---------------------------------------------------------------------------
__device__ __forceinline__ void tdm_load_b_tile(const __bf16* src, int K,
                                                unsigned lds_off) {
  unsigned long long ga = (unsigned long long)(uintptr_t)src;  // byte address
  u32x4 g0;
  g0[0] = 1u;                                   // count=1, user descriptor
  g0[1] = lds_off;                              // lds_addr (bytes)
  g0[2] = (unsigned)(ga & 0xFFFFFFFFu);         // global_addr[31:0]
  g0[3] = (unsigned)((ga >> 32) & 0x01FFFFFFu)  // global_addr[56:32]
          | (2u << 30);                         // type = 2 ("image")
  i32x8 g1;
  g1[0] = 1 << 16;                              // data_size = 1 -> 2 bytes
  g1[1] = (int)(((unsigned)K & 0xFFFFu) << 16); // tensor_dim0[15:0] @ bits 63:48
  g1[2] = (int)(1024u << 16);                   // tensor_dim1[15:0] @ bits 95:80
  g1[3] = 32 << 16;                             // tile_dim0 = 32   @ bits 127:112
  g1[4] = 64;                                   // tile_dim1 = 64   @ bits 143:128
  g1[5] = K;                                    // tensor_dim0_stride[31:0]
  g1[6] = 0;                                    // stride hi / dim1_stride lo
  g1[7] = 0;
  i32x4 z4 = {0, 0, 0, 0};                      // groups 2/3 unused (2D tile)
  i32x8 z8 = {0, 0, 0, 0, 0, 0, 0, 0};
  __builtin_amdgcn_tensor_load_to_lds(g0, g1, z4, z4, z8, 0);
}

// ---------------------------------------------------------------------------
// Elementwise f32 -> bf16 cast
// ---------------------------------------------------------------------------
__global__ __launch_bounds__(256) void cast_bf16_k(const float* __restrict__ in,
                                                   __bf16* __restrict__ out, int n) {
  int i = blockIdx.x * 256 + threadIdx.x;
  if (i < n) out[i] = (__bf16)in[i];
}

// f32 (1024x1024) -> bf16 transposed (so GEMM B-fragments are contiguous)
__global__ __launch_bounds__(256) void cast_transpose_k(const float* __restrict__ in,
                                                        __bf16* __restrict__ out) {
  int i = blockIdx.x * 256 + threadIdx.x;   // i = r*1024 + c
  int c = i & 1023;
  int r = i >> 10;
  out[(size_t)c * HID + r] = (__bf16)in[i];
}

// ---------------------------------------------------------------------------
// bf16 GEMM: C[M,N] = A[M,K] * Bt[N,K]^T, fp32 accumulate.
// B tile (64 x 32) staged into LDS by wave 0 via TDM, double-buffered across
// K-steps; all 8 waves read B fragments from LDS (ds_load) -> 8x less vmem.
// MODE 0: bf16 row-major, MODE 1: bf16 transposed (C^T), MODE 2: f32.
// Block = 256 threads (8 waves); wave tile = 16(M) x 64(N); K-step 32.
// ---------------------------------------------------------------------------
template <int MODE>
__global__ __launch_bounds__(256) void gemm_bf16_k(const __bf16* __restrict__ A,
                                                   const __bf16* __restrict__ Bt,
                                                   void* __restrict__ Cout,
                                                   int M, int N, int K, float scale) {
  __shared__ __bf16 ldsB[2][64][32];            // 2 x 4 KB double buffer

  const int lane = threadIdx.x & 31;
  const int wave = threadIdx.x >> 5;
  const int m0 = blockIdx.x * 128 + wave * 16;
  const int n0 = blockIdx.y * 64;
  const int r  = lane & 15;
  const int hi = lane >> 4;

  // Prologue: wave 0 kicks off the DMA for the first B tile.
  if (wave == 0) {
    tdm_load_b_tile(Bt + (size_t)n0 * K, K,
                    (unsigned)(uintptr_t)&ldsB[0][0][0]);
  }

  v8f acc[4] = {};
  // A layout: lane r holds row m0+r; K elems hi*16..hi*16+15 contiguous
  const __bf16* aRow = A + (size_t)(m0 + r) * K + hi * 16;

  for (int k0 = 0; k0 < K; k0 += 32) {
    const int buf = (k0 >> 5) & 1;
    __syncthreads();   // everyone done reading the buffer we are about to fill
    if (wave == 0) {
      if (k0 + 32 < K) {
        tdm_load_b_tile(Bt + (size_t)n0 * K + (k0 + 32), K,
                        (unsigned)(uintptr_t)&ldsB[buf ^ 1][0][0]);
        __builtin_amdgcn_s_wait_tensorcnt(1);   // current tile landed
      } else {
        __builtin_amdgcn_s_wait_tensorcnt(0);
      }
    }
    __syncthreads();   // publish current buffer to all waves

    __builtin_prefetch(aRow + k0 + 128, 0, 0);
    v8bf alo = *(const v8bf*)(aRow + k0);
    v8bf ahi = *(const v8bf*)(aRow + k0 + 8);
    v16bf af;
#pragma unroll
    for (int i = 0; i < 8; ++i) { af[i] = alo[i]; af[i + 8] = ahi[i]; }

    // B layout: lane r = column n; K elems hi*16..hi*16+15 contiguous in LDS
#pragma unroll
    for (int s = 0; s < 4; ++s) {
      v16bf bfv = *(const v16bf*)&ldsB[buf][s * 16 + r][hi * 16];
      acc[s] = wmma_bf16(af, bfv, acc[s]);
    }
  }

  // C/D layout: element e of lane -> row m0 + e + hi*8, col n0 + s*16 + r
#pragma unroll
  for (int s = 0; s < 4; ++s) {
#pragma unroll
    for (int e = 0; e < 8; ++e) {
      const int m = m0 + e + hi * 8;
      const int n = n0 + s * 16 + r;
      const float v = acc[s][e] * scale;
      if (MODE == 0)      ((__bf16*)Cout)[(size_t)m * N + n] = (__bf16)v;
      else if (MODE == 1) ((__bf16*)Cout)[(size_t)n * M + m] = (__bf16)v;
      else                ((float*)Cout)[(size_t)m * N + n] = v;
    }
  }
}

// ---------------------------------------------------------------------------
// NA2D flash attention.
// Q, K: bf16 [4096, 1024] ((x*64+y)*1024 + h*64 + d); Q pre-scaled by D^-0.5.
// Vt: bf16 [1024, 4096] transposed: Vt[(h*64+d)*4096 + x*64+y].
// O:  bf16 [4096, 1024].
// Grid (NH, XD), block 128 = 4 waves; wave w owns queries (x, y0=w*16 .. +15).
// Streams over 32 x-offsets and 32-key y-chunks with online softmax.
// ---------------------------------------------------------------------------
__global__ __launch_bounds__(128) void na2d_attn_k(const __bf16* __restrict__ Q,
                                                   const __bf16* __restrict__ Km,
                                                   const __bf16* __restrict__ Vt,
                                                   __bf16* __restrict__ O) {
  __shared__ __bf16 ldsP[4][16][32];

  const int lane = threadIdx.x & 31;
  const int wave = threadIdx.x >> 5;
  const int h  = blockIdx.x;
  const int x  = blockIdx.y;
  const int y0 = wave * 16;
  const int r  = lane & 15;
  const int hi = lane >> 4;

  // --- load Q A-fragments (16 queries x 64 d, split into two K=32 frags) ---
  const __bf16* qbase = Q + (size_t)(x * YD + y0 + r) * HID + h * DH + hi * 16;
  v16bf aq0, aq1;
  {
    v8bf l0 = *(const v8bf*)(qbase);
    v8bf h0 = *(const v8bf*)(qbase + 8);
    v8bf l1 = *(const v8bf*)(qbase + 32);
    v8bf h1 = *(const v8bf*)(qbase + 40);
#pragma unroll
    for (int i = 0; i < 8; ++i) {
      aq0[i] = l0[i]; aq0[i + 8] = h0[i];
      aq1[i] = l1[i]; aq1[i + 8] = h1[i];
    }
  }

  const int sx  = imin(imax(x - 15, 0), 32);
  // Align union start to 16 so all V loads stay 32B-aligned; extra keys masked.
  const int yLo = imin(imax(y0 - 15, 0), 32) & ~15;
  const int yHi = imin(y0, 32) + 32;
  const int nch = (yHi - yLo + 31) >> 5;

  float mrow[8], lrow[8];
#pragma unroll
  for (int i = 0; i < 8; ++i) { mrow[i] = -1e30f; lrow[i] = 0.0f; }
  v8f o[4] = {};

  for (int di = 0; di < 32; ++di) {
    const int kx = sx + di;
    for (int ch = 0; ch < nch; ++ch) {
      const int kyBase = yLo + ch * 32;

      // ----- scores: S(16x32) = Q(16x64) . K^T, two 16x16 WMMA tiles -----
      v8f s0 = {}, s1 = {};
      {
        const int ky = imin(kyBase + r, 63);
        const __bf16* kb = Km + (size_t)(kx * YD + ky) * HID + h * DH + hi * 16;
        v16bf b0 = *(const v16bf*)(kb);
        v16bf b1 = *(const v16bf*)(kb + 32);
        s0 = wmma_bf16(aq0, b0, s0);
        s0 = wmma_bf16(aq1, b1, s0);
      }
      {
        const int ky = imin(kyBase + 16 + r, 63);
        const __bf16* kb = Km + (size_t)(kx * YD + ky) * HID + h * DH + hi * 16;
        v16bf b0 = *(const v16bf*)(kb);
        v16bf b1 = *(const v16bf*)(kb + 32);
        s1 = wmma_bf16(aq0, b0, s1);
        s1 = wmma_bf16(aq1, b1, s1);
      }

      // ----- mask + online softmax (rows live across 16-lane groups) -----
#pragma unroll
      for (int e = 0; e < 8; ++e) {
        const int qy  = y0 + e + hi * 8;
        const int syq = imin(imax(qy - 15, 0), 32);
        const int ky0 = kyBase + r;
        const int ky1 = kyBase + 16 + r;
        float v0 = (ky0 >= syq && ky0 < syq + 32) ? s0[e] : -1e30f;
        float v1 = (ky1 >= syq && ky1 < syq + 32) ? s1[e] : -1e30f;
        float mx = fmaxf(v0, v1);
#pragma unroll
        for (int off = 1; off < 16; off <<= 1)
          mx = fmaxf(mx, __shfl_xor(mx, off, 32));
        const float mn = fmaxf(mrow[e], mx);
        const float sc = __expf(mrow[e] - mn);
        mrow[e] = mn;
        const float p0 = __expf(v0 - mn);
        const float p1 = __expf(v1 - mn);
        float ps = p0 + p1;
#pragma unroll
        for (int off = 1; off < 16; off <<= 1)
          ps += __shfl_xor(ps, off, 32);
        lrow[e] = lrow[e] * sc + ps;
#pragma unroll
        for (int dt = 0; dt < 4; ++dt) o[dt][e] *= sc;
        // stash P (C-layout) into LDS for layout transpose
        ldsP[wave][e + hi * 8][r]      = (__bf16)p0;
        ldsP[wave][e + hi * 8][16 + r] = (__bf16)p1;
      }

      // wave-local LDS visibility (no cross-wave sharing -> no barrier)
      asm volatile("s_wait_dscnt 0" ::: "memory");

      // ----- reload P in A-fragment layout (16 rows x 32 keys) -----
      v16bf ap;
#pragma unroll
      for (int i = 0; i < 8; ++i) {
        ap[i]     = ldsP[wave][r][hi * 16 + i];
        ap[i + 8] = ldsP[wave][r][hi * 16 + 8 + i];
      }

      // ----- O += P(16x32) . V(32x64); V keys contiguous in Vt rows -----
      const int kyv = imin(kyBase + hi * 16, 48);  // masked P zeroes cover clamp
#pragma unroll
      for (int dt = 0; dt < 4; ++dt) {
        const __bf16* vb =
            Vt + (size_t)(h * DH + dt * 16 + r) * QLEN + kx * YD + kyv;
        v16bf bv = *(const v16bf*)(vb);
        o[dt] = wmma_bf16(ap, bv, o[dt]);
      }
    }
  }

  // ----- normalize and store O (bf16) -----
#pragma unroll
  for (int dt = 0; dt < 4; ++dt) {
#pragma unroll
    for (int e = 0; e < 8; ++e) {
      const int m = e + hi * 8;
      const float v = o[dt][e] / lrow[e];
      O[(size_t)(x * YD + y0 + m) * HID + h * DH + dt * 16 + r] = (__bf16)v;
    }
  }
}

// ---------------------------------------------------------------------------
// Launcher
// ---------------------------------------------------------------------------
extern "C" void kernel_launch(void* const* d_in, const int* in_sizes, int n_in,
                              void* d_out, int out_size, void* d_ws, size_t ws_size,
                              hipStream_t stream) {
  (void)in_sizes; (void)n_in; (void)out_size; (void)ws_size;
  const float* hs = (const float*)d_in[0];
  const float* wq = (const float*)d_in[1];
  const float* wk = (const float*)d_in[2];
  const float* wv = (const float*)d_in[3];
  const float* wo = (const float*)d_in[4];
  float* out = (float*)d_out;

  char* ws = (char*)d_ws;
  size_t off = 0;
  __bf16* hsb = (__bf16*)(ws + off); off += (size_t)QLEN * HID * 2;   // 8 MB
  __bf16* wqt = (__bf16*)(ws + off); off += (size_t)HID * HID * 2;    // 2 MB
  __bf16* wkt = (__bf16*)(ws + off); off += (size_t)HID * HID * 2;
  __bf16* wvt = (__bf16*)(ws + off); off += (size_t)HID * HID * 2;
  __bf16* wot = (__bf16*)(ws + off); off += (size_t)HID * HID * 2;
  __bf16* Qb  = (__bf16*)(ws + off); off += (size_t)QLEN * HID * 2;
  __bf16* Kb  = (__bf16*)(ws + off); off += (size_t)QLEN * HID * 2;
  __bf16* Vt  = (__bf16*)(ws + off); off += (size_t)QLEN * HID * 2;   // transposed
  __bf16* Ob  = (__bf16*)(ws + off); off += (size_t)QLEN * HID * 2;

  // 1) precision casts
  cast_bf16_k<<<(QLEN * HID) / 256, 256, 0, stream>>>(hs, hsb, QLEN * HID);
  cast_transpose_k<<<(HID * HID) / 256, 256, 0, stream>>>(wq, wqt);
  cast_transpose_k<<<(HID * HID) / 256, 256, 0, stream>>>(wk, wkt);
  cast_transpose_k<<<(HID * HID) / 256, 256, 0, stream>>>(wv, wvt);
  cast_transpose_k<<<(HID * HID) / 256, 256, 0, stream>>>(wo, wot);

  // 2) projections (Q pre-scaled by D^-0.5 = 1/8); V stored transposed
  dim3 ggrid(QLEN / 128, HID / 64);
  gemm_bf16_k<0><<<ggrid, 256, 0, stream>>>(hsb, wqt, (void*)Qb, QLEN, HID, HID, 0.125f);
  gemm_bf16_k<0><<<ggrid, 256, 0, stream>>>(hsb, wkt, (void*)Kb, QLEN, HID, HID, 1.0f);
  gemm_bf16_k<1><<<ggrid, 256, 0, stream>>>(hsb, wvt, (void*)Vt, QLEN, HID, HID, 1.0f);

  // 3) neighborhood attention (flash-style streaming)
  na2d_attn_k<<<dim3(NH, XD), 128, 0, stream>>>(Qb, Kb, Vt, Ob);

  // 4) output projection, fp32 result
  gemm_bf16_k<2><<<ggrid, 256, 0, stream>>>(Ob, wot, (void*)out, QLEN, HID, HID, 1.0f);
}
